// DFD_43327630082789
// MI455X (gfx1250) — matrix operand order repaired and checked by reading, not compiled, not run on hardware
//
#include <hip/hip_runtime.h>

typedef __attribute__((ext_vector_type(16))) _Float16 v16h;
typedef __attribute__((ext_vector_type(8)))  _Float16 v8h;
typedef __attribute__((ext_vector_type(8)))  float    v8f;
typedef __attribute__((ext_vector_type(2)))  float    f2v;

#define B_        8
#define FEAT_     256
#define NTOT_     21824
#define XS_       264          // padded LDS row stride (halfs), 528B = 33*16B
#define HS_       264
#define TILE_     64
#define NTHREADS_ 128

// f16 weight workspace layout (element offsets)
#define W16_CLS_WH 0
#define W16_BOX_WH 65536
#define W16_CLS_WO 131072
#define W16_BOX_WO 151552      // padded to 16x256 (rows 4..15 zero)
#define W16_TOTAL  155648

// ---------------- weight conversion pre-kernel ----------------
__global__ __launch_bounds__(256) void cvt_weights_kernel(
    const float* __restrict__ cls_wh, const float* __restrict__ cls_wo,
    const float* __restrict__ box_wh, const float* __restrict__ box_wo,
    _Float16* __restrict__ w16)
{
  int i = blockIdx.x * 256 + threadIdx.x;
  if (i >= W16_TOTAL) return;
  float v;
  if (i < W16_BOX_WH)        v = cls_wh[i];
  else if (i < W16_CLS_WO)   v = box_wh[i - W16_BOX_WH];
  else if (i < W16_BOX_WO)   v = cls_wo[i - W16_CLS_WO];
  else { int j = i - W16_BOX_WO; v = (j < 1024) ? box_wo[j] : 0.0f; }
  w16[i] = (_Float16)v;
}

// ---------------- fragment helpers ----------------
__device__ __forceinline__ v8f wmma_f16(v16h a, v16h b, v8f c) {
  // (neg_a, A, neg_b, B, c_mod, C, reuse_a, reuse_b)
  return __builtin_amdgcn_wmma_f32_16x16x32_f16(false, a, false, b, (short)0, c,
                                                false, false);
}

// A (16xK f16) fragment from LDS row-major [row][k], stride halfs.
// lanes 0-15: halfs = K kk*32+{0..7, 16..23}; lanes 16-31: +8 / +24.
__device__ __forceinline__ v16h ldsA(const _Float16* base, int lane, int kk, int stride) {
  int row = lane & 15;
  int off = kk * 32 + ((lane >> 4) << 3);
  union { v16h v; v8h h[2]; } u;
  u.h[0] = *(const v8h*)(base + row * stride + off);
  u.h[1] = *(const v8h*)(base + row * stride + off + 16);
  return u.v;
}

// B (Kx16 f16) fragment from global row-major weight [n][256].
// lane holds column n = nbase + lane%16; lanes 0-15: K kk*32+{0..7, 16..23}; lanes 16-31: +8 / +24.
__device__ __forceinline__ v16h gldB(const _Float16* w, int nbase, int lane, int kk) {
  int row = nbase + (lane & 15);
  int off = kk * 32 + ((lane >> 4) << 3);
  const v8h* p = (const v8h*)(w + row * 256 + off);
  union { v16h v; v8h h[2]; } u;
  u.h[0] = p[0];
  u.h[1] = p[2];
  return u.v;
}

// hidden layer: [16 tok x 256] = X(16x256) * Wh^T, + bias, ReLU -> f16 LDS
__device__ __forceinline__ void hidden_layer(const _Float16* xbase,
                                             const _Float16* wh,
                                             const float* __restrict__ bh,
                                             _Float16* hdst, int lane)
{
  v16h af[8];
#pragma unroll
  for (int kk = 0; kk < 8; ++kk) af[kk] = ldsA(xbase, lane, kk, XS_);
  for (int j = 0; j < 16; ++j) {
    v8f c = {};
#pragma unroll
    for (int kk = 0; kk < 8; ++kk)
      c = wmma_f16(af[kk], gldB(wh, j * 16, lane, kk), c);
    int col = j * 16 + (lane & 15);
    float bb = bh[col];
#pragma unroll
    for (int r = 0; r < 8; ++r) {
      float v = c[r] + bb;
      v = v > 0.0f ? v : 0.0f;                 // ReLU
      int row = r + ((lane >> 4) << 3);        // D layout: lanes16-31 -> rows 8..15
      hdst[row * HS_ + col] = (_Float16)v;
    }
  }
}

// ---------------- fused detection head ----------------
__global__ __launch_bounds__(NTHREADS_) void det_head_kernel(
    const float* __restrict__ fm0, const float* __restrict__ fm1,
    const float* __restrict__ fm2, const float* __restrict__ fm3,
    const float* __restrict__ fm4,
    const _Float16* __restrict__ w16,
    const float* __restrict__ cls_bh, const float* __restrict__ cls_bo,
    const float* __restrict__ box_bh, const float* __restrict__ box_bo,
    float* __restrict__ out)
{
  __shared__ __align__(16) _Float16 sm[TILE_ * XS_ + 4 * 16 * HS_]; // 67584 B

  const int tid  = threadIdx.x;
  const int lane = tid & 31;
  const int wave = tid >> 5;

  const int g  = blockIdx.x;
  const int t0 = g * TILE_;            // global token index (b*NTOT + n0)
  const int b  = t0 / NTOT_;
  const int n0 = t0 - b * NTOT_;

  // tile lies entirely within one pyramid level (all boundaries are %64==0)
  const float* fmL; int HWl, lvlOff, lsh;
  if (n0 < 16384)      { fmL = fm0; HWl = 16384; lvlOff = 0;     lsh = 7; }
  else if (n0 < 20480) { fmL = fm1; HWl = 4096;  lvlOff = 16384; lsh = 6; }
  else if (n0 < 21504) { fmL = fm2; HWl = 1024;  lvlOff = 20480; lsh = 5; }
  else if (n0 < 21760) { fmL = fm3; HWl = 256;   lvlOff = 21504; lsh = 4; }
  else                 { fmL = fm4; HWl = 64;    lvlOff = 21760; lsh = 3; }
  const int   Wl   = 1 << lsh;
  const float rW   = 1.0f / (float)Wl;
  const int   pos0 = n0 - lvlOff;

  const float* src = fmL + (size_t)b * FEAT_ * HWl + pos0;

  // ---- stage X tile [64 tok x 256 feat] as f16, row-major, padded stride ----
  // Feature maps are read exactly once per launch -> non-temporal loads keep
  // the 178MB stream from churning WGP$/L2 (protect weight residency).
  for (int it = 0; it < 64; ++it) {
    int idx = it * NTHREADS_ + tid;        // 0..8191 (256 feats * 32 token-pairs)
    int f   = idx >> 5;
    int i   = (idx & 31) << 1;
    const f2v d = __builtin_nontemporal_load((const f2v*)(src + (size_t)f * HWl + i));
    sm[i * XS_ + f]       = (_Float16)d.x;
    sm[(i + 1) * XS_ + f] = (_Float16)d.y;
  }
  __syncthreads();

  const _Float16* xbase = sm + wave * 16 * XS_;
  _Float16*       hbase = sm + TILE_ * XS_ + wave * 16 * HS_;

  // ================= cls head =================
  hidden_layer(xbase, w16 + W16_CLS_WH, cls_bh, hbase, lane);
  __syncthreads();
  {
    v16h hf[8];
#pragma unroll
    for (int kk = 0; kk < 8; ++kk) hf[kk] = ldsA(hbase, lane, kk, HS_);
    for (int j = 0; j < 5; ++j) {          // 80 classes = 5 N-tiles
      v8f c = {};
#pragma unroll
      for (int kk = 0; kk < 8; ++kk)
        c = wmma_f16(hf[kk], gldB(w16 + W16_CLS_WO, j * 16, lane, kk), c);
      int col = j * 16 + (lane & 15);
      float bb = cls_bo[col];
#pragma unroll
      for (int r = 0; r < 8; ++r) {
        int m  = r + ((lane >> 4) << 3);
        int tg = t0 + wave * 16 + m;
        __builtin_nontemporal_store(c[r] + bb, &out[(size_t)tg * 84 + col]);
      }
    }
  }
  __syncthreads();

  // ================= box head =================
  hidden_layer(xbase, w16 + W16_BOX_WH, box_bh, hbase, lane);
  __syncthreads();
  {
    v16h hf[8];
#pragma unroll
    for (int kk = 0; kk < 8; ++kk) hf[kk] = ldsA(hbase, lane, kk, HS_);
    v8f c = {};
#pragma unroll
    for (int kk = 0; kk < 8; ++kk)
      c = wmma_f16(hf[kk], gldB(w16 + W16_BOX_WO, 0, lane, kk), c);
    // divergence only AFTER the last WMMA (EXEC must be all-1s for WMMA)
    int col = lane & 15;
    if (col < 4) {
      float bb = box_bo[col];
#pragma unroll
      for (int r = 0; r < 8; ++r) {
        int m   = r + ((lane >> 4) << 3);
        int n   = n0 + wave * 16 + m;
        int pos = n - lvlOff;
        int px  = pos & (Wl - 1);
        int py  = pos >> lsh;
        float t = tanhf(c[r] + bb);
        float val;
        if (col == 0)      val = ((float)px + 0.5f) * rW + 0.1f * t;  // cx
        else if (col == 1) val = ((float)py + 0.5f) * rW + 0.1f * t;  // cy
        else               val = rW * exp2f(t);                        // w,h: (1/W)*2^d
        int tg = t0 + wave * 16 + m;
        __builtin_nontemporal_store(val, &out[(size_t)tg * 84 + 80 + col]);
      }
    }
  }
}

// ---------------- launch ----------------
extern "C" void kernel_launch(void* const* d_in, const int* in_sizes, int n_in,
                              void* d_out, int out_size, void* d_ws, size_t ws_size,
                              hipStream_t stream) {
  const float* fm0    = (const float*)d_in[0];
  const float* fm1    = (const float*)d_in[1];
  const float* fm2    = (const float*)d_in[2];
  const float* fm3    = (const float*)d_in[3];
  const float* fm4    = (const float*)d_in[4];
  const float* cls_Wh = (const float*)d_in[5];
  const float* cls_bh = (const float*)d_in[6];
  const float* cls_Wo = (const float*)d_in[7];
  const float* cls_bo = (const float*)d_in[8];
  const float* box_Wh = (const float*)d_in[9];
  const float* box_bh = (const float*)d_in[10];
  const float* box_Wo = (const float*)d_in[11];
  const float* box_bo = (const float*)d_in[12];
  _Float16* w16 = (_Float16*)d_ws;          // 311,296 B of scratch used
  float* out = (float*)d_out;

  cvt_weights_kernel<<<(W16_TOTAL + 255) / 256, 256, 0, stream>>>(
      cls_Wh, cls_Wo, box_Wh, box_Wo, w16);

  const int nblocks = (B_ * NTOT_) / TILE_;  // 2728
  det_head_kernel<<<nblocks, NTHREADS_, 0, stream>>>(
      fm0, fm1, fm2, fm3, fm4, w16, cls_bh, cls_bo, box_bh, box_bo, out);
}
